// BeaconGPT_58136677318864
// MI455X (gfx1250) — matrix operand order repaired and checked by reading, compiled, not run on hardware
//
#include <hip/hip_runtime.h>
#include <hip/hip_bf16.h>

// ---------------------------------------------------------------------------
// Types for CDNA5 WMMA
// ---------------------------------------------------------------------------
typedef __attribute__((ext_vector_type(16))) __bf16        v16bf;
typedef __attribute__((ext_vector_type(8)))  float         v8f;
typedef __attribute__((ext_vector_type(8)))  unsigned int  v8u;

union Frag {
    v8u   u;
    v16bf b;
};

__device__ inline unsigned short f2bf(float f) {
    unsigned int u = __float_as_uint(f);
    unsigned int r = u + 0x7FFFu + ((u >> 16) & 1u); // round-to-nearest-even
    return (unsigned short)(r >> 16);
}

// ---------------------------------------------------------------------------
// CDNA5 async global->LDS copy (16 bytes per lane), tracked by ASYNCcnt.
// VDST VGPR carries the per-lane LDS byte offset (low 32 bits of the generic
// LDS pointer are the LDS offset on AMDGPU); VADDR is the 64-bit global addr.
// ---------------------------------------------------------------------------
__device__ inline void async_copy16(unsigned lds_off, const void* gptr) {
    asm volatile("global_load_async_to_lds_b128 %0, %1, off"
                 :
                 : "v"(lds_off), "v"((unsigned long long)(uintptr_t)gptr)
                 : "memory");
}
__device__ inline void wait_async0() {
    asm volatile("s_wait_asynccnt 0" ::: "memory");
}

// ---------------------------------------------------------------------------
// Model constants
// ---------------------------------------------------------------------------
#define SEQ      2048
#define HIDDEN   1024
#define NHEAD    16
#define HDIM     64
#define NLAYER   8
#define VOCAB    50272
#define EPSF     1e-5f

// GEMM tiling
#define BM  128
#define BN  64
#define BK  32
#define LDA 40   // padded LDS stride (halves) for A tile rows [m][k]
#define LDB 40   // padded LDS stride (halves) for B tile rows [n][k]

// ---------------------------------------------------------------------------
// WMMA GEMM:  C[M,N] = A[M,K](bf16) * op(B)   with per-mode epilogue
//   BNK=true : B stored [N,K] (row-major N-rows), TB = float or ushort(bf16)
//   BNK=false: B stored [K,N] (bf16 only)
// mode 0: C = alpha*acc (fp32)     mode 1: C += alpha*acc (fp32)
// mode 2: Cb = bf16(acc)           mode 3: Cb = bf16(relu(acc)^2)
// M is covered exactly by gridDim.y*BM (always 2048 here).
// ---------------------------------------------------------------------------
template <typename TB, bool BNK>
__global__ void __launch_bounds__(256)
gemm_wmma(const unsigned short* __restrict__ A, int lda,
          const TB* __restrict__ B, int ldb,
          float* __restrict__ C, unsigned short* __restrict__ Cb, int ldc,
          int N, int K, int mode, float alpha)
{
    __shared__ unsigned short sA[BM * LDA];
    __shared__ unsigned short sB[BN * LDB];

    const int m0   = blockIdx.y * BM;
    const int n0   = blockIdx.x * BN;
    const int tid  = threadIdx.x;
    const int lane = tid & 31;
    const int wid  = tid >> 5;          // 8 waves
    const int wm   = (wid >> 1) * 32;   // wave row offset   (4 x 32 = 128)
    const int wn   = (wid & 1) * 32;    // wave col offset   (2 x 32 =  64)

    v8f acc[2][2] = {};

    for (int kt = 0; kt < K; kt += BK) {
        // ---- stage A tile (bf16 [BM x BK]) via async global->LDS ----
        {
            int row = tid >> 1;
            int seg = tid & 1;
            const unsigned short* src = A + (size_t)(m0 + row) * lda + kt + seg * 16;
            unsigned dst = (unsigned)(uintptr_t)&sA[row * LDA + seg * 16];
            async_copy16(dst,      src);
            async_copy16(dst + 16, src + 8);
        }
        // ---- stage B tile into sB[n][k] (bf16) ----
        if constexpr (BNK) {
            int n   = tid >> 2;
            int seg = tid & 3;
            int gn  = n0 + n;
            if constexpr (sizeof(TB) == 4) {
                // fp32 weights: load, convert to bf16 in-core, ds_store
                uint4 w = make_uint4(0u, 0u, 0u, 0u);
                if (gn < N) {
                    const float* src = (const float*)B + (size_t)gn * ldb + kt + seg * 8;
                    if (kt + BK < K) __builtin_prefetch((const char*)(src + BK), 0, 0);
                    float4 f0 = *(const float4*)(src);
                    float4 f1 = *(const float4*)(src + 4);
                    w.x = (unsigned)f2bf(f0.x) | ((unsigned)f2bf(f0.y) << 16);
                    w.y = (unsigned)f2bf(f0.z) | ((unsigned)f2bf(f0.w) << 16);
                    w.z = (unsigned)f2bf(f1.x) | ((unsigned)f2bf(f1.y) << 16);
                    w.w = (unsigned)f2bf(f1.z) | ((unsigned)f2bf(f1.w) << 16);
                }
                *(uint4*)&sB[n * LDB + seg * 8] = w;
            } else {
                // bf16 [N,K]: straight 16B copy -> async global->LDS
                if (gn < N) {
                    const unsigned short* src = (const unsigned short*)B +
                                                (size_t)gn * ldb + kt + seg * 8;
                    unsigned dst = (unsigned)(uintptr_t)&sB[n * LDB + seg * 8];
                    async_copy16(dst, src);
                }
                // gn >= N lanes: LDS left stale; those output columns are
                // never stored (n < N guard in epilogue).
            }
        } else {
            // B stored [K,N] bf16 (e.g. V matrix): transpose while staging
            int n  = tid & 63;
            int kb = (tid >> 6) * 8;
            int gn = n0 + n;
            const unsigned short* Bs = (const unsigned short*)B;
            #pragma unroll
            for (int r = 0; r < 8; r++) {
                int k = kb + r;
                sB[n * LDB + k] =
                    (gn < N) ? Bs[(size_t)(kt + k) * ldb + gn] : (unsigned short)0;
            }
        }
        wait_async0();      // ASYNCcnt -> 0 for this wave's async tile copies
        __syncthreads();

        // ---- load fragments per documented CDNA5 VGPR layouts ----
        Frag a[2], b[2];
        {
            int g    = lane >> 4;
            int mrow = lane & 15;
            #pragma unroll
            for (int ti = 0; ti < 2; ti++) {
                const unsigned short* p = &sA[(wm + ti * 16 + mrow) * LDA];
                #pragma unroll
                for (int i = 0; i < 8; i++) {
                    int kb2 = ((i >= 4) ? 16 : 0) + g * 8 + (i & 3) * 2;
                    a[ti].u[i] = *(const unsigned int*)(p + kb2);
                }
            }
            int ncol = lane & 15;
            int kg   = lane >> 4;
            #pragma unroll
            for (int tj = 0; tj < 2; tj++) {
                const unsigned short* p = &sB[(wn + tj * 16 + ncol) * LDB + kg * 8];
                #pragma unroll
                for (int i = 0; i < 8; i++)
                    b[tj].u[i] = *(const unsigned int*)(p + ((i >= 4) ? 16 : 0) + (i & 3) * 2);
            }
        }

        // ---- 4 x v_wmma_f32_16x16x32_bf16 ----
        #pragma unroll
        for (int ti = 0; ti < 2; ti++)
            #pragma unroll
            for (int tj = 0; tj < 2; tj++)
                acc[ti][tj] = __builtin_amdgcn_wmma_f32_16x16x32_bf16(
                    false, a[ti].b, false, b[tj].b,
                    (short)0, acc[ti][tj], false, false);
        __syncthreads();
    }

    // ---- epilogue (D layout: m = r + 8*(lane>>4), n = lane&15) ----
    int ncol  = lane & 15;
    int rmoff = (lane >> 4) * 8;
    #pragma unroll
    for (int ti = 0; ti < 2; ti++) {
        #pragma unroll
        for (int tj = 0; tj < 2; tj++) {
            int n = n0 + wn + tj * 16 + ncol;
            if (n < N) {
                #pragma unroll
                for (int r = 0; r < 8; r++) {
                    int m = m0 + wm + ti * 16 + r + rmoff;
                    size_t idx = (size_t)m * ldc + n;
                    float v = acc[ti][tj][r];
                    if (mode == 0)      C[idx]  = alpha * v;
                    else if (mode == 1) C[idx] += alpha * v;
                    else if (mode == 2) Cb[idx] = f2bf(v);
                    else { float t = fmaxf(v, 0.f); Cb[idx] = f2bf(t * t); }
                }
            }
        }
    }
}

// ---------------------------------------------------------------------------
// Embedding gather + RMSNorm -> fp32 x
// ---------------------------------------------------------------------------
__global__ void __launch_bounds__(256)
embed_rmsnorm(const int* __restrict__ ids, const float* __restrict__ wte,
              float* __restrict__ x)
{
    int row = blockIdx.x;
    const float* src = wte + (size_t)ids[row] * HIDDEN;
    __shared__ float red[256];
    float v[4]; float ss = 0.f;
    #pragma unroll
    for (int i = 0; i < 4; i++) { v[i] = src[threadIdx.x + 256 * i]; ss += v[i] * v[i]; }
    red[threadIdx.x] = ss; __syncthreads();
    for (int off = 128; off > 0; off >>= 1) {
        if (threadIdx.x < off) red[threadIdx.x] += red[threadIdx.x + off];
        __syncthreads();
    }
    float sc = rsqrtf(red[0] * (1.f / HIDDEN) + EPSF);
    float* dst = x + (size_t)row * HIDDEN;
    #pragma unroll
    for (int i = 0; i < 4; i++) dst[threadIdx.x + 256 * i] = v[i] * sc;
}

// ---------------------------------------------------------------------------
// RMSNorm fp32 row -> bf16 row
// ---------------------------------------------------------------------------
__global__ void __launch_bounds__(256)
rmsnorm_bf16(const float* __restrict__ x, unsigned short* __restrict__ o)
{
    int row = blockIdx.x;
    const float* src = x + (size_t)row * HIDDEN;
    __shared__ float red[256];
    float v[4]; float ss = 0.f;
    #pragma unroll
    for (int i = 0; i < 4; i++) { v[i] = src[threadIdx.x + 256 * i]; ss += v[i] * v[i]; }
    red[threadIdx.x] = ss; __syncthreads();
    for (int off = 128; off > 0; off >>= 1) {
        if (threadIdx.x < off) red[threadIdx.x] += red[threadIdx.x + off];
        __syncthreads();
    }
    float sc = rsqrtf(red[0] * (1.f / HIDDEN) + EPSF);
    unsigned short* dst = o + (size_t)row * HIDDEN;
    #pragma unroll
    for (int i = 0; i < 4; i++) dst[threadIdx.x + 256 * i] = f2bf(v[i] * sc);
}

// ---------------------------------------------------------------------------
// QKV post: per (pos, head) rmsnorm(q), rmsnorm(k), RoPE(q,k), convert v.
// Writes head-major bf16 tensors [head][pos][HDIM].
// ---------------------------------------------------------------------------
__global__ void __launch_bounds__(64)
qkv_post(const float* __restrict__ qkv,
         unsigned short* __restrict__ qh,
         unsigned short* __restrict__ kh,
         unsigned short* __restrict__ vh)
{
    int pos = blockIdx.x, h = blockIdx.y, t = threadIdx.x;
    const float* base = qkv + (size_t)pos * (3 * HIDDEN) + h * HDIM;
    float q = base[t];
    float k = base[HIDDEN + t];
    float v = base[2 * HIDDEN + t];

    __shared__ float red[2][64];
    __shared__ float qn[64], kn[64];
    red[0][t] = q * q; red[1][t] = k * k; __syncthreads();
    for (int off = 32; off > 0; off >>= 1) {
        if (t < off) { red[0][t] += red[0][t + off]; red[1][t] += red[1][t + off]; }
        __syncthreads();
    }
    float qs = rsqrtf(red[0][0] * (1.f / HDIM) + EPSF);
    float ks = rsqrtf(red[1][0] * (1.f / HDIM) + EPSF);
    qn[t] = q * qs; kn[t] = k * ks; __syncthreads();

    size_t ho = ((size_t)h * SEQ + pos) * HDIM;
    vh[ho + t] = f2bf(v);

    if (t < 32) {
        int i = t;
        // inv_freq = 10000^(-2i/64) = exp(-2i/64 * ln(10000))
        float inv = __expf(-(float)(2 * i) * (9.210340371976184f / 64.f));
        float ang = (float)pos * inv;
        float c = __cosf(ang), s = __sinf(ang);
        float qe = qn[2 * i], qo = qn[2 * i + 1];
        qh[ho + 2 * i]     = f2bf(qo * c - qe * s);
        qh[ho + 2 * i + 1] = f2bf(qo * s + qe * c);
        float ke = kn[2 * i], ko = kn[2 * i + 1];
        kh[ho + 2 * i]     = f2bf(ko * c - ke * s);
        kh[ho + 2 * i + 1] = f2bf(ko * s + ke * c);
    }
}

// ---------------------------------------------------------------------------
// Causal softmax: fp32 scores row -> bf16 probs row (zeros above diagonal)
// ---------------------------------------------------------------------------
__global__ void __launch_bounds__(256)
softmax_causal(const float* __restrict__ S, unsigned short* __restrict__ P)
{
    int row = blockIdx.x;
    const float* s = S + (size_t)row * SEQ;
    unsigned short* pr = P + (size_t)row * SEQ;
    int valid = row + 1;
    __shared__ float red[256];

    float mx = -3.4e38f;
    for (int j = threadIdx.x; j < valid; j += 256) mx = fmaxf(mx, s[j]);
    red[threadIdx.x] = mx; __syncthreads();
    for (int off = 128; off > 0; off >>= 1) {
        if (threadIdx.x < off)
            red[threadIdx.x] = fmaxf(red[threadIdx.x], red[threadIdx.x + off]);
        __syncthreads();
    }
    mx = red[0]; __syncthreads();

    float sum = 0.f;
    for (int j = threadIdx.x; j < valid; j += 256) sum += __expf(s[j] - mx);
    red[threadIdx.x] = sum; __syncthreads();
    for (int off = 128; off > 0; off >>= 1) {
        if (threadIdx.x < off) red[threadIdx.x] += red[threadIdx.x + off];
        __syncthreads();
    }
    float invs = 1.f / red[0];

    for (int j = threadIdx.x; j < SEQ; j += 256) {
        float v = (j < valid) ? __expf(s[j] - mx) * invs : 0.f;
        pr[j] = f2bf(v);
    }
}

// ---------------------------------------------------------------------------
// Host orchestration
// ---------------------------------------------------------------------------
extern "C" void kernel_launch(void* const* d_in, const int* in_sizes, int n_in,
                              void* d_out, int out_size, void* d_ws, size_t ws_size,
                              hipStream_t stream)
{
    (void)in_sizes; (void)n_in; (void)out_size; (void)ws_size;

    const int*   ids  = (const int*)  d_in[0];
    const float* wte  = (const float*)d_in[1];
    const float* wqkv = (const float*)d_in[2];
    const float* wo   = (const float*)d_in[3];
    const float* w1   = (const float*)d_in[4];
    const float* w2   = (const float*)d_in[5];
    const float* lm   = (const float*)d_in[6];
    float*       out  = (float*)d_out;

    // workspace carve (all 256B aligned; total ~92 MB)
    char* p = (char*)d_ws;
    auto carve = [&](size_t bytes) -> void* {
        void* r = (void*)p;
        p += (bytes + 255) & ~(size_t)255;
        return r;
    };
    float*          x      = (float*)         carve((size_t)SEQ * HIDDEN * 4);
    unsigned short* hbf    = (unsigned short*)carve((size_t)SEQ * HIDDEN * 2);
    float*          qkv    = (float*)         carve((size_t)SEQ * 3 * HIDDEN * 4);
    unsigned short* qh     = (unsigned short*)carve((size_t)NHEAD * SEQ * HDIM * 2);
    unsigned short* kh     = (unsigned short*)carve((size_t)NHEAD * SEQ * HDIM * 2);
    unsigned short* vh     = (unsigned short*)carve((size_t)NHEAD * SEQ * HDIM * 2);
    float*          scores = (float*)         carve((size_t)SEQ * SEQ * 4);
    unsigned short* probs  = (unsigned short*)carve((size_t)SEQ * SEQ * 2);
    unsigned short* attnbf = (unsigned short*)carve((size_t)SEQ * HIDDEN * 2);
    unsigned short* ubuf   = (unsigned short*)carve((size_t)SEQ * 4 * HIDDEN * 2);

    const int MT = SEQ / BM; // 16 M-tiles

    embed_rmsnorm<<<SEQ, 256, 0, stream>>>(ids, wte, x);

    for (int l = 0; l < NLAYER; l++) {
        const float* wqkv_l = wqkv + (size_t)l * 3 * HIDDEN * HIDDEN;
        const float* wo_l   = wo   + (size_t)l * HIDDEN * HIDDEN;
        const float* w1_l   = w1   + (size_t)l * 4 * HIDDEN * HIDDEN;
        const float* w2_l   = w2   + (size_t)l * HIDDEN * 4 * HIDDEN;

        // h = rmsnorm(x) (bf16)
        rmsnorm_bf16<<<SEQ, 256, 0, stream>>>(x, hbf);

        // qkv = h @ wqkv^T  (fp32 out)
        gemm_wmma<float, true><<<dim3((3 * HIDDEN) / BN, MT), 256, 0, stream>>>(
            hbf, HIDDEN, wqkv_l, HIDDEN, qkv, nullptr, 3 * HIDDEN,
            3 * HIDDEN, HIDDEN, 0, 1.0f);

        // per-head rmsnorm + RoPE, v convert (head-major bf16)
        qkv_post<<<dim3(SEQ, NHEAD), 64, 0, stream>>>(qkv, qh, kh, vh);

        // attention, one head at a time (scores/probs scratch reused)
        for (int h = 0; h < NHEAD; h++) {
            const unsigned short* qp = qh + (size_t)h * SEQ * HDIM;
            const unsigned short* kp = kh + (size_t)h * SEQ * HDIM;
            const unsigned short* vp = vh + (size_t)h * SEQ * HDIM;

            // scores = (q @ k^T) / 8
            gemm_wmma<unsigned short, true><<<dim3(SEQ / BN, MT), 256, 0, stream>>>(
                qp, HDIM, kp, HDIM, scores, nullptr, SEQ,
                SEQ, HDIM, 0, 0.125f);

            softmax_causal<<<SEQ, 256, 0, stream>>>(scores, probs);

            // attn_out[:, h*64 : h*64+64] = probs @ v   (bf16 out)
            gemm_wmma<unsigned short, false><<<dim3(HDIM / BN, MT), 256, 0, stream>>>(
                probs, SEQ, vp, HDIM, nullptr, attnbf + h * HDIM, HIDDEN,
                HDIM, SEQ, 2, 1.0f);
        }

        // x += attn_out @ wo^T
        gemm_wmma<float, true><<<dim3(HIDDEN / BN, MT), 256, 0, stream>>>(
            attnbf, HIDDEN, wo_l, HIDDEN, x, nullptr, HIDDEN,
            HIDDEN, HIDDEN, 1, 1.0f);

        // h2 = rmsnorm(x)
        rmsnorm_bf16<<<SEQ, 256, 0, stream>>>(x, hbf);

        // u = relu(h2 @ w1^T)^2  (bf16 out)
        gemm_wmma<float, true><<<dim3((4 * HIDDEN) / BN, MT), 256, 0, stream>>>(
            hbf, HIDDEN, w1_l, HIDDEN, nullptr, ubuf, 4 * HIDDEN,
            4 * HIDDEN, HIDDEN, 3, 1.0f);

        // x += u @ w2^T
        gemm_wmma<float, true><<<dim3(HIDDEN / BN, MT), 256, 0, stream>>>(
            ubuf, 4 * HIDDEN, w2_l, 4 * HIDDEN, x, nullptr, HIDDEN,
            HIDDEN, 4 * HIDDEN, 1, 1.0f);
    }

    // final rmsnorm + lm_head
    rmsnorm_bf16<<<SEQ, 256, 0, stream>>>(x, hbf);
    gemm_wmma<float, true><<<dim3((VOCAB + BN - 1) / BN, MT), 256, 0, stream>>>(
        hbf, HIDDEN, lm, HIDDEN, out, nullptr, VOCAB,
        VOCAB, HIDDEN, 0, 1.0f);
}
